// BasicBlock_53730040873264
// MI455X (gfx1250) — 8-bit matrix operand order repaired and checked by reading, compiled, not run on hardware
//
#include <hip/hip_runtime.h>
#include <cstdint>

typedef int v8i __attribute__((ext_vector_type(8)));

#define NB   32
#define CCH  64
#define HH   112
#define WWD  112
#define HWD  (HH*WWD)
#define QLEV 15.0f

#define WQ1_OFF   0
#define WQ2_OFF   36864
#define BNA1_OFF  73728
#define BNB1_OFF  (BNA1_OFF+256)
#define BNA2_OFF  (BNA1_OFF+512)
#define BNB2_OFF  (BNA1_OFF+768)
#define ACT1_OFF  131072
#define ACT_BYTES ((long long)NB*CCH*HH*WWD)
#define ACT2_OFF  (ACT1_OFF + ACT_BYTES)

#if __has_builtin(__builtin_amdgcn_global_load_async_to_lds_b128)
#define HAVE_ASYNC_LDS 1
#else
#define HAVE_ASYNC_LDS 0
#endif

typedef int gvec4i __attribute__((vector_size(16)));
typedef __attribute__((address_space(1))) gvec4i as1_v4i;
typedef __attribute__((address_space(3))) gvec4i as3_v4i;

__device__ __forceinline__ as1_v4i* to_as1_v4(const void* p) {
    return (as1_v4i*)(uintptr_t)p;
}
__device__ __forceinline__ as3_v4i* to_as3_v4(void* p) {
    return (as3_v4i*)(unsigned)(uintptr_t)p;
}
__device__ __forceinline__ void wait_asynccnt0() {
#if __has_builtin(__builtin_amdgcn_s_wait_asynccnt)
    __builtin_amdgcn_s_wait_asynccnt(0);
#else
    asm volatile("s_wait_asynccnt 0" ::: "memory");
#endif
}

__global__ void prep_kernel(const float* __restrict__ w1, const float* __restrict__ w2,
                            const float* g1, const float* b1, const float* m1, const float* v1,
                            const float* g2, const float* b2, const float* m2, const float* v2,
                            unsigned char* __restrict__ ws)
{
    __shared__ float red[256];
    const int tid = threadIdx.x;

    if (blockIdx.x == 2) {
        if (tid < 64) {
            float inv = rsqrtf(v1[tid] + 1e-5f);
            float A   = g1[tid] * inv;
            ((float*)(ws + BNA1_OFF))[tid] = A;
            ((float*)(ws + BNB1_OFF))[tid] = b1[tid] - m1[tid] * A;
        } else if (tid < 128) {
            int c = tid - 64;
            float inv = rsqrtf(v2[c] + 1e-5f);
            float A   = g2[c] * inv;
            ((float*)(ws + BNA2_OFF))[c] = A;
            ((float*)(ws + BNB2_OFF))[c] = b2[c] - m2[c] * A;
        }
        return;
    }

    const float* w = (blockIdx.x == 0) ? w1 : w2;
    signed char* wq = (signed char*)(ws + ((blockIdx.x == 0) ? WQ1_OFF : WQ2_OFF));
    const int NW = 64 * 64 * 9;

    float mx = 0.0f;
    for (int i = tid; i < NW; i += 256)
        mx = fmaxf(mx, fabsf(tanhf(w[i])));
    red[tid] = mx;
    __syncthreads();
    for (int s = 128; s > 0; s >>= 1) {
        if (tid < s) red[tid] = fmaxf(red[tid], red[tid + s]);
        __syncthreads();
    }
    const float inv_mx = 1.0f / red[0];

    for (int i = tid; i < NW; i += 256) {
        float t = tanhf(w[i]) * inv_mx;
        float u = (t + 1.0f) * 0.5f;
        int   q = 2 * (int)rintf(u * QLEV) - 15;
        int dx = i % 3;
        int dy = (i / 3) % 3;
        int ic = (i / 9) % 64;
        int o  = i / 576;
        int mt = o >> 4, m = o & 15;
        wq[(((mt * 9 + dy * 3 + dx) * 16 + m) * 64) + ic] = (signed char)q;
    }
}

__global__ void quant_in_kernel(const float* __restrict__ x, const float* __restrict__ alpha,
                                unsigned char* __restrict__ actq)
{
    long long idx = (long long)blockIdx.x * blockDim.x + threadIdx.x;
    const long long total = (long long)NB * CCH * HWD;
    if (idx >= total) return;
    float a     = *alpha;
    float inv_s = QLEV / a;
    int w = (int)(idx % WWD);
    int h = (int)((idx / WWD) % HH);
    int c = (int)((idx / HWD) % CCH);
    int n = (int)(idx / ((long long)HWD * CCH));
    float xv = x[idx];
    xv = fminf(fmaxf(xv, 0.0f), a);
    int q = (int)rintf(xv * inv_s);
    actq[(((long long)(n * HH + h) * WWD + w) * CCH) + c] = (unsigned char)q;
}

template <int EPI>
__global__ __launch_bounds__(128, 1)
void conv_kernel(const unsigned char* __restrict__ actq, const unsigned char* __restrict__ wq,
                 const float* __restrict__ bnA, const float* __restrict__ bnB,
                 const float* __restrict__ alphaIn, const float* __restrict__ alphaOut,
                 const float* __restrict__ xres,
                 unsigned char* __restrict__ out_q, float* __restrict__ out_f)
{
    __shared__ int4 lds4[3 * 20 * 4];

    const int tid   = threadIdx.x;
    const int lane  = tid & 31;
    const int wid   = tid >> 5;
    const int wt    = blockIdx.x;
    const int hc    = blockIdx.y;
    const int n     = blockIdx.z;
    const int w0    = wt * 16;
    const int m0    = wid * 16;
    const int col   = lane & 15;
    const int khalf = lane >> 4;

    const float aIn   = *alphaIn;
    const float sconv = aIn / (QLEV * QLEV);
    float qscaleOut = 0.0f;
    if (EPI == 0) qscaleOut = QLEV / (*alphaOut);

    v8i afrag[9];
    {
        const int koff = khalf * 16;
        const unsigned char* wbase = wq + (((wid * 9) * 16 + col) * 64);
#pragma unroll
        for (int kk = 0; kk < 9; ++kk) {
            const unsigned char* wp = wbase + kk * (16 * 64);
            uint2 p0 = *(const uint2*)(wp + 0  + koff);
            uint2 p1 = *(const uint2*)(wp + 8  + koff);
            uint2 p2 = *(const uint2*)(wp + 32 + koff);
            uint2 p3 = *(const uint2*)(wp + 40 + koff);
            v8i a;
            a[0] = (int)p0.x; a[1] = (int)p0.y; a[2] = (int)p1.x; a[3] = (int)p1.y;
            a[4] = (int)p2.x; a[5] = (int)p2.y; a[6] = (int)p3.x; a[7] = (int)p3.y;
            afrag[kk] = a;
        }
    }

    const int chBase = m0 + khalf * 8;
    float4 A0 = *(const float4*)(bnA + chBase);
    float4 A1 = *(const float4*)(bnA + chBase + 4);
    float4 B0 = *(const float4*)(bnB + chBase);
    float4 B1 = *(const float4*)(bnB + chBase + 4);
    float abn[8] = {A0.x, A0.y, A0.z, A0.w, A1.x, A1.y, A1.z, A1.w};
    float bbn[8] = {B0.x, B0.y, B0.z, B0.w, B1.x, B1.y, B1.z, B1.w};

    for (int hi = 0; hi < 28; ++hi) {
        const int h = hc * 28 + hi;

        __syncthreads();
        for (int t = tid; t < 216; t += 128) {
            int e = t >> 2, q = t & 3;
            int row = e / 18, ci = e % 18;
            int hh = h - 1 + row;
            int ww = w0 - 1 + ci;
            int  li = (row * 20 + ci) * 4 + q;
            if (hh >= 0 && hh < HH && ww >= 0 && ww < WWD) {
                const unsigned char* gp =
                    actq + ((((long long)n * HH + hh) * WWD + ww) * CCH) + q * 16;
#if HAVE_ASYNC_LDS
                __builtin_amdgcn_global_load_async_to_lds_b128(
                    to_as1_v4(gp), to_as3_v4(&lds4[li]),  0,  0);
#else
                lds4[li] = *(const int4*)gp;
#endif
                if (hh + 3 < HH) __builtin_prefetch(gp + 3LL * WWD * CCH, 0, 0);
            } else {
                lds4[li] = make_int4(0, 0, 0, 0);
            }
        }
#if HAVE_ASYNC_LDS
        wait_asynccnt0();
#endif
        __syncthreads();

        v8i bfrag[9];
#pragma unroll
        for (int dy = 0; dy < 3; ++dy) {
#pragma unroll
            for (int dx = 0; dx < 3; ++dx) {
                const int base = (dy * 20 + (col + dx)) * 4 + khalf;
                int4 lo  = lds4[base];
                int4 hi4 = lds4[base + 2];
                v8i b;
                b[0] = lo.x;  b[1] = lo.y;  b[2] = lo.z;  b[3] = lo.w;
                b[4] = hi4.x; b[5] = hi4.y; b[6] = hi4.z; b[7] = hi4.w;
                bfrag[dy * 3 + dx] = b;
            }
        }

        v8i acc = {};
#pragma unroll
        for (int s = 0; s < 9; ++s) {
            acc = __builtin_amdgcn_wmma_i32_16x16x64_iu8(
                       true, afrag[s],
                       false, bfrag[s],
                      acc,  false,  false);
        }

        if (EPI == 0) {
            unsigned int packed[2] = {0u, 0u};
#pragma unroll
            for (int r = 0; r < 8; ++r) {
                float y = (float)acc[r] * sconv;
                y = y * abn[r] + bbn[r];
                y = fmaxf(y, 0.0f);
                float qf = rintf(fminf(y * qscaleOut, QLEV));
                int q = (int)fmaxf(qf, 0.0f);
                packed[r >> 2] |= (unsigned int)(q & 0xFF) << ((r & 3) * 8);
            }
            long long sp = ((long long)n * HH + h) * WWD + (w0 + col);
            *(uint2*)(out_q + sp * CCH + chBase) = make_uint2(packed[0], packed[1]);
        } else {
#pragma unroll
            for (int r = 0; r < 8; ++r) {
                int ch = chBase + r;
                long long oaddr = (((long long)n * CCH + ch) * HH + h) * WWD + (w0 + col);
                float y = (float)acc[r] * sconv;
                y = y * abn[r] + bbn[r];
                y += xres[oaddr];
                out_f[oaddr] = fmaxf(y, 0.0f);
            }
        }
    }
}

extern "C" void kernel_launch(void* const* d_in, const int* in_sizes, int n_in,
                              void* d_out, int out_size, void* d_ws, size_t ws_size,
                              hipStream_t stream)
{
    const float* x      = (const float*)d_in[0];
    const float* w1     = (const float*)d_in[1];
    const float* alpha1 = (const float*)d_in[2];
    const float* g1     = (const float*)d_in[3];
    const float* b1     = (const float*)d_in[4];
    const float* m1     = (const float*)d_in[5];
    const float* v1     = (const float*)d_in[6];
    const float* w2     = (const float*)d_in[7];
    const float* alpha2 = (const float*)d_in[8];
    const float* g2     = (const float*)d_in[9];
    const float* b2     = (const float*)d_in[10];
    const float* m2     = (const float*)d_in[11];
    const float* v2     = (const float*)d_in[12];

    unsigned char* ws    = (unsigned char*)d_ws;
    unsigned char* wq1   = ws + WQ1_OFF;
    unsigned char* wq2   = ws + WQ2_OFF;
    const float*   bnA1  = (const float*)(ws + BNA1_OFF);
    const float*   bnB1  = (const float*)(ws + BNB1_OFF);
    const float*   bnA2  = (const float*)(ws + BNA2_OFF);
    const float*   bnB2  = (const float*)(ws + BNB2_OFF);
    unsigned char* act1q = ws + ACT1_OFF;
    unsigned char* act2q = ws + ACT2_OFF;

    prep_kernel<<<3, 256, 0, stream>>>(w1, w2, g1, b1, m1, v1, g2, b2, m2, v2, ws);

    long long total = (long long)NB * CCH * HWD;
    int qblocks = (int)((total + 255) / 256);
    quant_in_kernel<<<qblocks, 256, 0, stream>>>(x, alpha1, act1q);

    conv_kernel<0><<<dim3(7, 4, 32), 128, 0, stream>>>(
        act1q, wq1, bnA1, bnB1, alpha1, alpha2, nullptr, act2q, nullptr);

    conv_kernel<1><<<dim3(7, 4, 32), 128, 0, stream>>>(
        act2q, wq2, bnA2, bnB2, alpha2, alpha2, x, nullptr, (float*)d_out);
}
